// TSEncoder_84181359001806
// MI455X (gfx1250) — compile-verified, matrix operand order checked by reading, not run on hardware
//
#include <hip/hip_runtime.h>
#include <hip/hip_bf16.h>
#include <math.h>

// ---------------------------------------------------------------------------
// Types for CDNA5 WMMA (gfx1250, wave32)
// ---------------------------------------------------------------------------
typedef __attribute__((ext_vector_type(16))) __bf16 v16bf;
typedef __attribute__((ext_vector_type(8)))  float  v8f;

struct BF16x16 { uint4 lo, hi; };   // 32 bytes == v16bf

__device__ __forceinline__ unsigned short f2bfbits(float f) {
  unsigned int u = __builtin_bit_cast(unsigned int, f);
  unsigned int lsb = (u >> 16) & 1u;
  u += 0x7FFFu + lsb;               // round-to-nearest-even
  return (unsigned short)(u >> 16);
}

__device__ __forceinline__ float gelu_exact(float v) {
  return 0.5f * v * (1.0f + erff(v * 0.70710678118654752f));
}

// ---------------------------------------------------------------------------
// 0. zero small accumulator region
// ---------------------------------------------------------------------------
__global__ void zero_kernel(float* p) { p[threadIdx.x] = 0.0f; }

// ---------------------------------------------------------------------------
// 1. Weight prep: (Cout, Cin, ntaps) fp32 -> bf16 in WMMA A-fragment layout.
//    Conv reads: Wp[(((coT*ntaps + tap)*nchunk + chunk)*32 + lane)*16 + e]
//    lane: m = lane&15 (row M), half = lane>>4
//    element e -> K = chunk*32 + (e<8 ? half*8+e : 16+half*8+(e-8))
// ---------------------------------------------------------------------------
__global__ void prep_weights_kernel(const float* __restrict__ w,
                                    unsigned short* __restrict__ out,
                                    int Cout, int Cin, int ntaps) {
  int total = Cout * Cin * ntaps;
  int idx = blockIdx.x * blockDim.x + threadIdx.x;
  if (idx >= total) return;
  int e    = idx & 15;
  int lane = (idx >> 4) & 31;
  int rem  = idx >> 9;
  int nchunk = Cin >> 5;
  int chunk = rem % nchunk;  rem /= nchunk;
  int tap   = rem % ntaps;
  int coT   = rem / ntaps;
  int m = lane & 15, half = lane >> 4;
  int K  = chunk * 32 + (e < 8 ? half * 8 + e : 16 + half * 8 + (e - 8));
  int co = coT * 16 + m;
  out[idx] = f2bfbits(w[((size_t)co * Cin + K) * ntaps + tap]);
}

// ---------------------------------------------------------------------------
// 2. Sliding-window(5) linear + FC(64->256), fused.
//    Emits fp32 X [b][t][h] AND gelu(X) as bf16 (next conv's input).
// ---------------------------------------------------------------------------
__global__ __launch_bounds__(256)
void windowfc_kernel(const float* __restrict__ x, const float* __restrict__ wreg,
                     const float* __restrict__ breg, const float* __restrict__ wfc,
                     const float* __restrict__ bfc, float* __restrict__ out,
                     unsigned short* __restrict__ gout,
                     int T, int Tp) {
  __shared__ float sy[16][64];
  int tilesT = (Tp + 15) >> 4;
  int b  = blockIdx.x / tilesT;
  int t0 = (blockIdx.x % tilesT) << 4;
  int tid = threadIdx.x;
  float w0 = wreg[0], w1 = wreg[1], w2 = wreg[2], w3 = wreg[3], w4 = wreg[4];
  float br = breg[0];
  for (int i = tid; i < 16 * 64; i += 256) {
    int tt = i >> 6, d = i & 63;
    int t = t0 + tt;
    float v = 0.0f;
    if (t < Tp) {
      const float* xp = x + ((size_t)b * T + t) * 64 + d;
      v = xp[0] * w0 + xp[64] * w1 + xp[128] * w2 + xp[192] * w3 + xp[256] * w4 + br;
    }
    sy[tt][d] = v;
  }
  __syncthreads();
  int h = tid;                       // H == 256 == blockDim
  float wr[64];
  #pragma unroll
  for (int d = 0; d < 64; ++d) wr[d] = wfc[(size_t)h * 64 + d];
  float bh = bfc[h];
  for (int tt = 0; tt < 16; ++tt) {
    int t = t0 + tt;
    if (t >= Tp) break;
    float a = bh;
    #pragma unroll
    for (int d = 0; d < 64; ++d) a += sy[tt][d] * wr[d];
    size_t o = ((size_t)b * Tp + t) * 256 + h;
    out[o]  = a;
    gout[o] = f2bfbits(gelu_exact(a));
  }
}

// ---------------------------------------------------------------------------
// 3. plain fp32 -> bf16 pack (no GELU), for the 1x1 projection input
// ---------------------------------------------------------------------------
__global__ void pack_bf16_kernel(const float* __restrict__ in,
                                 unsigned short* __restrict__ out,
                                 long long n) {
  long long i = (long long)blockIdx.x * blockDim.x + threadIdx.x;
  if (i >= n) return;
  out[i] = f2bfbits(in[i]);
}

// ---------------------------------------------------------------------------
// 4. Dilated conv1d (same padding) as WMMA GEMM. One wave computes a
//    32(co) x 32(t) macro-tile: 2 A-frags x 2 B-frags -> 4 WMMAs per
//    (tap, chunk) step (~16 FLOP/B from L2). Fused epilogue can emit:
//      fout : fp32  acc + bias (+ residual)          [b][t][co]
//      gout : bf16  gelu(acc + bias (+ residual))    [b][t][co]
// ---------------------------------------------------------------------------
template<int NCHUNK, int NTAPS>
__global__ __launch_bounds__(32)
void conv_wmma_kernel(const unsigned short* __restrict__ G,
                      const unsigned short* __restrict__ Wp,
                      const float* __restrict__ bias,
                      const float* __restrict__ res,
                      float* __restrict__ fout,
                      unsigned short* __restrict__ gout,
                      int Tp, int Cout, int dil) {
  const int CIN = NCHUNK * 32;
  int lane = threadIdx.x;
  int half = lane >> 4, m16 = lane & 15;
  int tiles = (Tp + 31) >> 5;
  int b  = blockIdx.x / tiles;
  int t0 = (blockIdx.x % tiles) << 5;
  int coT0 = blockIdx.y << 1;                    // two 16-row co tiles
  const size_t coStride = (size_t)NTAPS * NCHUNK * 32 * 16;

  // prefetch this wave's weight stream (L2-resident across waves)
  __builtin_prefetch(Wp + (size_t)coT0 * coStride + (size_t)lane * 16, 0, 3);

  const v8f z8 = {0.f, 0.f, 0.f, 0.f, 0.f, 0.f, 0.f, 0.f};
  v8f a00 = z8, a01 = z8, a10 = z8, a11 = z8;    // [coSub][tSub]
  const uint4 z4 = make_uint4(0u, 0u, 0u, 0u);

  #pragma unroll
  for (int tap = 0; tap < NTAPS; ++tap) {
    int off = (tap - (NTAPS - 1) / 2) * dil;     // same padding
    int ta = t0 + m16 + off;
    int tb = ta + 16;
    bool ia = (ta >= 0) && (ta < Tp);
    bool ib = (tb >= 0) && (tb < Tp);
    const unsigned short* ga = G + (size_t)(b * Tp + (ia ? ta : 0)) * CIN;
    const unsigned short* gb = G + (size_t)(b * Tp + (ib ? tb : 0)) * CIN;
    #pragma unroll
    for (int c = 0; c < NCHUNK; ++c) {
      const unsigned short* ap0 =
          Wp + ((((size_t)coT0 * NTAPS + tap) * NCHUNK + c) * 32 + lane) * 16;
      const unsigned short* ap1 = ap0 + coStride;
      BF16x16 A0, A1, B0, B1;
      A0.lo = *(const uint4*)(ap0);      A0.hi = *(const uint4*)(ap0 + 8);
      A1.lo = *(const uint4*)(ap1);      A1.hi = *(const uint4*)(ap1 + 8);
      B0.lo = ia ? *(const uint4*)(ga + c * 32 + half * 8)      : z4;
      B0.hi = ia ? *(const uint4*)(ga + c * 32 + 16 + half * 8) : z4;
      B1.lo = ib ? *(const uint4*)(gb + c * 32 + half * 8)      : z4;
      B1.hi = ib ? *(const uint4*)(gb + c * 32 + 16 + half * 8) : z4;
      v16bf av0 = __builtin_bit_cast(v16bf, A0);
      v16bf av1 = __builtin_bit_cast(v16bf, A1);
      v16bf bv0 = __builtin_bit_cast(v16bf, B0);
      v16bf bv1 = __builtin_bit_cast(v16bf, B1);
      a00 = __builtin_amdgcn_wmma_f32_16x16x32_bf16(false, av0, false, bv0, (short)0, a00, false, false);
      a01 = __builtin_amdgcn_wmma_f32_16x16x32_bf16(false, av0, false, bv1, (short)0, a01, false, false);
      a10 = __builtin_amdgcn_wmma_f32_16x16x32_bf16(false, av1, false, bv0, (short)0, a10, false, false);
      a11 = __builtin_amdgcn_wmma_f32_16x16x32_bf16(false, av1, false, bv1, (short)0, a11, false, false);
    }
  }

  auto store_tile = [&](const v8f& acc, int coT, int t) {
    if (t >= Tp) return;
    int co = (coT << 4) + half * 8;               // D rows M = j + 8*half
    size_t o = (size_t)(b * Tp + t) * Cout + co;
    float4 b0 = *(const float4*)(bias + co);
    float4 b1 = *(const float4*)(bias + co + 4);
    float v[8];
    v[0] = acc[0] + b0.x; v[1] = acc[1] + b0.y;
    v[2] = acc[2] + b0.z; v[3] = acc[3] + b0.w;
    v[4] = acc[4] + b1.x; v[5] = acc[5] + b1.y;
    v[6] = acc[6] + b1.z; v[7] = acc[7] + b1.w;
    if (res) {
      float4 r0 = *(const float4*)(res + o);
      float4 r1 = *(const float4*)(res + o + 4);
      v[0] += r0.x; v[1] += r0.y; v[2] += r0.z; v[3] += r0.w;
      v[4] += r1.x; v[5] += r1.y; v[6] += r1.z; v[7] += r1.w;
    }
    if (fout) {
      *(float4*)(fout + o)     = make_float4(v[0], v[1], v[2], v[3]);
      *(float4*)(fout + o + 4) = make_float4(v[4], v[5], v[6], v[7]);
    }
    if (gout) {
      unsigned int p0 = (unsigned)f2bfbits(gelu_exact(v[0])) | ((unsigned)f2bfbits(gelu_exact(v[1])) << 16);
      unsigned int p1 = (unsigned)f2bfbits(gelu_exact(v[2])) | ((unsigned)f2bfbits(gelu_exact(v[3])) << 16);
      unsigned int p2 = (unsigned)f2bfbits(gelu_exact(v[4])) | ((unsigned)f2bfbits(gelu_exact(v[5])) << 16);
      unsigned int p3 = (unsigned)f2bfbits(gelu_exact(v[6])) | ((unsigned)f2bfbits(gelu_exact(v[7])) << 16);
      *(uint4*)(gout + o) = make_uint4(p0, p1, p2, p3);
    }
  };
  store_tile(a00, coT0,     t0 + m16);
  store_tile(a01, coT0,     t0 + 16 + m16);
  store_tile(a10, coT0 + 1, t0 + m16);
  store_tile(a11, coT0 + 1, t0 + 16 + m16);
}

// ---------------------------------------------------------------------------
// 5. loss part 1: trace(H H^T) = sum(x_out[0]^2)
// ---------------------------------------------------------------------------
__global__ __launch_bounds__(256)
void sumsq_kernel(const float* __restrict__ x, long long n, float* __restrict__ accum) {
  __shared__ float red[256];
  float s = 0.f;
  long long stride = (long long)gridDim.x * 256;
  for (long long i = (long long)blockIdx.x * 256 + threadIdx.x; i < n; i += stride) {
    float v = x[i]; s += v * v;
  }
  red[threadIdx.x] = s; __syncthreads();
  for (int w = 128; w > 0; w >>= 1) {
    if (threadIdx.x < w) red[threadIdx.x] += red[threadIdx.x + w];
    __syncthreads();
  }
  if (threadIdx.x == 0) atomicAdd(accum, red[0]);
}

// ---------------------------------------------------------------------------
// 6. loss part 2: orthogonal init (pseudo-normal + MGS QR, r_kk>0 so the
//    sign fix is a no-op) then trace(Q^T H H^T Q) = sum_k || X q_k ||^2.
// ---------------------------------------------------------------------------
__device__ __forceinline__ unsigned int splitmix(unsigned int x) {
  x += 0x9e3779b9u; x ^= x >> 16; x *= 0x85ebca6bu;
  x ^= x >> 13; x *= 0xc2b2ae35u; x ^= x >> 16; return x;
}

__global__ __launch_bounds__(512)
void kmeans_loss_kernel(const float* __restrict__ xout0, const float* __restrict__ sumsq,
                        float* __restrict__ loss_out, int Tp, int O) {
  const int K = 6;
  __shared__ float A[320 * 6];
  __shared__ float red[512];
  int tid = threadIdx.x;
  for (int i = tid; i < O * K; i += 512) {
    unsigned int h1 = splitmix(0x01234567u + (unsigned)i * 2u);
    unsigned int h2 = splitmix(0x89abcdefu + (unsigned)i * 2u + 1u);
    float u1 = (h1 >> 8) * (1.0f / 16777216.0f) + 1e-7f;
    float u2 = (h2 >> 8) * (1.0f / 16777216.0f);
    A[i] = sqrtf(-2.0f * logf(u1)) * cosf(6.2831853f * u2);
  }
  __syncthreads();
  for (int k = 0; k < K; ++k) {                       // modified Gram-Schmidt
    for (int j = 0; j < k; ++j) {
      red[tid] = (tid < O) ? A[tid * K + j] * A[tid * K + k] : 0.f;
      __syncthreads();
      for (int w = 256; w > 0; w >>= 1) { if (tid < w) red[tid] += red[tid + w]; __syncthreads(); }
      float dot = red[0]; __syncthreads();
      if (tid < O) A[tid * K + k] -= dot * A[tid * K + j];
      __syncthreads();
    }
    red[tid] = (tid < O) ? A[tid * K + k] * A[tid * K + k] : 0.f;
    __syncthreads();
    for (int w = 256; w > 0; w >>= 1) { if (tid < w) red[tid] += red[tid + w]; __syncthreads(); }
    float inv = rsqrtf(fmaxf(red[0], 1e-20f)); __syncthreads();
    if (tid < O) A[tid * K + k] *= inv;
    __syncthreads();
  }
  float local = 0.f;
  for (int t = tid; t < Tp; t += 512) {
    const float* row = xout0 + (size_t)t * O;
    float s[K] = {0.f, 0.f, 0.f, 0.f, 0.f, 0.f};
    for (int o = 0; o < O; ++o) {
      float v = row[o];
      #pragma unroll
      for (int k = 0; k < K; ++k) s[k] += v * A[o * K + k];
    }
    #pragma unroll
    for (int k = 0; k < K; ++k) local += s[k] * s[k];
  }
  red[tid] = local; __syncthreads();
  for (int w = 256; w > 0; w >>= 1) { if (tid < w) red[tid] += red[tid + w]; __syncthreads(); }
  if (tid == 0) loss_out[0] = 0.05f * (sumsq[0] - red[0]);
}

// ---------------------------------------------------------------------------
// Launch
// ---------------------------------------------------------------------------
extern "C" void kernel_launch(void* const* d_in, const int* in_sizes, int n_in,
                              void* d_out, int out_size, void* d_ws, size_t ws_size,
                              hipStream_t stream) {
  (void)in_sizes; (void)n_in; (void)out_size; (void)ws_size;
  const int B = 16, T = 2048, O = 320, Tp = T - 5 + 1;     // 2044
  const int tilesT  = (Tp + 15) / 16;                       // 128 (stem)
  const int tilesT2 = (Tp + 31) / 32;                       // 64  (conv macro-tiles)

  const float* x    = (const float*)d_in[0];
  const float* wreg = (const float*)d_in[1];
  const float* breg = (const float*)d_in[2];
  const float* wfc  = (const float*)d_in[3];
  const float* bfc  = (const float*)d_in[4];
  struct Blk { const float *w1, *b1, *w2, *b2, *pw, *pb; };
  Blk blk[11];
  int p = 5;
  for (int i = 0; i < 10; ++i) {
    blk[i] = { (const float*)d_in[p], (const float*)d_in[p+1],
               (const float*)d_in[p+2], (const float*)d_in[p+3], nullptr, nullptr };
    p += 4;
  }
  blk[10] = { (const float*)d_in[p],   (const float*)d_in[p+1],
              (const float*)d_in[p+2], (const float*)d_in[p+3],
              (const float*)d_in[p+4], (const float*)d_in[p+5] };

  // ---- workspace carve-up (256B aligned regions) ----
  char* ws = (char*)d_ws;
  size_t off = 0;
  float* scal = (float*)ws; off += 256;
  unsigned short* w1p[11]; unsigned short* w2p[11]; unsigned short* pwp;
  auto alloc_us = [&](size_t elems) {
    unsigned short* r = (unsigned short*)(ws + off);
    off += ((elems * 2 + 255) / 256) * 256;
    return r;
  };
  for (int i = 0; i < 11; ++i) {
    int cin = 256, cout = (i == 10) ? 320 : 256;
    w1p[i] = alloc_us((size_t)cout * cin * 3);
    w2p[i] = alloc_us((size_t)cout * cout * 3);
  }
  pwp = alloc_us((size_t)320 * 256);
  const size_t actE = (size_t)B * Tp * 320;                 // max-channel extent
  unsigned short* Ga = (unsigned short*)(ws + off); off += ((actE * 2 + 255) / 256) * 256;
  unsigned short* Gb = (unsigned short*)(ws + off); off += ((actE * 2 + 255) / 256) * 256;
  float* Xa = (float*)(ws + off); off += actE * 4;
  float* Xb = (float*)(ws + off); off += actE * 4;

  // ---- per-call init + weight prep ----
  zero_kernel<<<1, 64, 0, stream>>>(scal);
  for (int i = 0; i < 11; ++i) {
    int cin = 256, cout = (i == 10) ? 320 : 256;
    int n1 = cout * cin * 3, n2 = cout * cout * 3;
    prep_weights_kernel<<<(n1 + 255) / 256, 256, 0, stream>>>(blk[i].w1, w1p[i], cout, cin, 3);
    prep_weights_kernel<<<(n2 + 255) / 256, 256, 0, stream>>>(blk[i].w2, w2p[i], cout, cout, 3);
  }
  prep_weights_kernel<<<(320 * 256 + 255) / 256, 256, 0, stream>>>(blk[10].pw, pwp, 320, 256, 1);

  // ---- stem: window(5) + FC -> Xa (fp32) and Ga (gelu bf16) ----
  windowfc_kernel<<<B * tilesT, 256, 0, stream>>>(x, wreg, breg, wfc, bfc, Xa, Ga, T, Tp);

  // ---- 10 residual blocks, C = 256 ----
  float* cur = Xa; float* nxt = Xb;
  for (int i = 0; i < 10; ++i) {
    int dil = 1 << i;
    // h1 = conv1(gelu(X)); emit gelu(h1) bf16 only
    conv_wmma_kernel<8, 3><<<dim3(B * tilesT2, 8), 32, 0, stream>>>(
        Ga, w1p[i], blk[i].b1, nullptr, nullptr, Gb, Tp, 256, dil);
    // X' = conv2(gelu(h1)) + X; emit X' fp32 and gelu(X') bf16
    conv_wmma_kernel<8, 3><<<dim3(B * tilesT2, 8), 32, 0, stream>>>(
        Gb, w2p[i], blk[i].b2, /*res=*/cur, nxt, Ga, Tp, 256, dil);
    float* tmp = cur; cur = nxt; nxt = tmp;                 // cur ends back at Xa
  }

  // ---- final block (256 -> 320, dil = 1024), residual via 1x1 projection ----
  float* xout = (float*)d_out;                              // [b][t][320]
  const long long n256 = (long long)B * Tp * 256;
  // R = pw @ X + pb -> Xb   (needs non-GELU bf16 of X)
  pack_bf16_kernel<<<(unsigned)((n256 + 255) / 256), 256, 0, stream>>>(cur, Gb, n256);
  conv_wmma_kernel<8, 1><<<dim3(B * tilesT2, 10), 32, 0, stream>>>(
      Gb, pwp, blk[10].pb, nullptr, nxt, nullptr, Tp, 320, 1);
  // h1 = conv1(gelu(X)); emit gelu(h1) bf16 (320 ch)
  conv_wmma_kernel<8, 3><<<dim3(B * tilesT2, 10), 32, 0, stream>>>(
      Ga, w1p[10], blk[10].b1, nullptr, nullptr, Gb, Tp, 320, 1024);
  // x_out = conv2(gelu(h1)) + R  -> straight into d_out
  conv_wmma_kernel<10, 3><<<dim3(B * tilesT2, 10), 32, 0, stream>>>(
      Gb, w2p[10], blk[10].b2, /*res=*/nxt, xout, nullptr, Tp, 320, 1024);

  // ---- k-means loss on x_out[0] ----
  sumsq_kernel<<<256, 256, 0, stream>>>(xout, (long long)Tp * O, scal);
  kmeans_loss_kernel<<<1, 512, 0, stream>>>(xout, scal, xout + (size_t)B * Tp * O, Tp, O);
}
